// HWNN_4355096839081
// MI455X (gfx1250) — matrix operand order repaired and checked by reading, not compiled, not run on hardware
//
#include <hip/hip_runtime.h>
#include <cstddef>

// ---------------------------------------------------------------------------
// HWNN hypergraph polynomial filter for MI455X (gfx1250, wave32, WMMA).
//   out = (p0 I + p1 T + p2 T^2) diag(d) (p3 I + p4 T^T + p5 T^T^2) X W
// 4 big [N,N]x[N,F] GEMMs (fp32 -> fp16 tiles, fp32 WMMA accumulation),
// 2 elementwise combines, 1 small GEMM with W. Double-buffered LDS,
// register-transposed staging (b64 DS stores), one barrier per K-step.
// ---------------------------------------------------------------------------

typedef _Float16 f16;
typedef __attribute__((ext_vector_type(4)))  float    v4f;
typedef __attribute__((ext_vector_type(4)))  _Float16 v4h;
typedef __attribute__((ext_vector_type(8)))  _Float16 v8h;
typedef __attribute__((ext_vector_type(16))) _Float16 v16h;
typedef __attribute__((ext_vector_type(8)))  float    v8f;

#define BM   128   // output rows per block
#define BN   128   // output cols per block
#define BK   32    // K-step (matches 16x16x32 f16 WMMA)
#define LDSK 40    // padded LDS row stride in f16 (80 B; 8B/16B aligned chunks)

// C[M,Ncols] = op(A)[M,K] * B[K,Ncols]; op = transpose if TRANSA.
// A indexed: TRANSA ? A[k*lda + m] : A[m*lda + k]. All dims multiples of tile.
template <bool TRANSA>
__global__ __launch_bounds__(256)
void gemm_f16_wmma(const float* __restrict__ A, const float* __restrict__ B,
                   float* __restrict__ C, int M, int Ncols, int K,
                   int lda, int ldb, int ldc)
{
    __shared__ f16 As[2][BM][LDSK];   // As[buf][m][k]
    __shared__ f16 Bs[2][BN][LDSK];   // Bs[buf][n][k] (B tile transposed)

    const int tid   = threadIdx.x;
    const int lane  = tid & 31;
    const int wave  = tid >> 5;         // 0..7
    const int wm    = wave >> 1;        // 0..3 -> 32-row strip
    const int wn    = wave & 1;         // 0..1 -> 64-col strip
    const int l16   = lane & 15;
    const int khalf = lane >> 4;        // 0/1

    const int m0 = blockIdx.x * BM;
    const int n0 = blockIdx.y * BN;

    // micro-tile coords for the 32(k) x 128(col) staging loads
    const int kb  = (tid >> 5) * 4;     // 0,4,...,28   (k-row group)
    const int cb  = (tid & 31) * 4;     // 0,4,...,124  (column group)

    // ---- staging helpers: 4 coalesced b128 loads, register transpose, 4 b64 DS stores
    auto loadA = [&](int kbase, v4f ra[4]) {
        if (TRANSA) {
            // memory tile: A[kbase+kb+j][m0+cb .. +3]
#pragma unroll
            for (int j = 0; j < 4; ++j)
                ra[j] = *(const v4f*)(A + (size_t)(kbase + kb + j) * lda + m0 + cb);
        } else {
            // memory tile: A[m0+mm][kbase + c4*4 .. +3], 128 rows x 8 float4
#pragma unroll
            for (int j = 0; j < 4; ++j) {
                int idx = tid + j * 256;          // 0..1023
                int mm  = idx >> 3;               // 0..127
                int c4  = idx & 7;                // 0..7
                ra[j] = *(const v4f*)(A + (size_t)(m0 + mm) * lda + kbase + c4 * 4);
            }
        }
    };
    auto loadB = [&](int kbase, v4f rb[4]) {
#pragma unroll
        for (int j = 0; j < 4; ++j)
            rb[j] = *(const v4f*)(B + (size_t)(kbase + kb + j) * ldb + n0 + cb);
    };
    auto storeTile = [&](const v4f ra[4], const v4f rb[4], int buf) {
        if (TRANSA) {
            // As[m][k]: m = cb+c, k = kb..kb+3 packed in one b64 store
#pragma unroll
            for (int c = 0; c < 4; ++c) {
                v4h h;
#pragma unroll
                for (int j = 0; j < 4; ++j) h[j] = (f16)ra[j][c];
                *(v4h*)&As[buf][cb + c][kb] = h;
            }
        } else {
#pragma unroll
            for (int j = 0; j < 4; ++j) {
                int idx = tid + j * 256;
                int mm  = idx >> 3;
                int c4  = idx & 7;
                v4h h;
#pragma unroll
                for (int c = 0; c < 4; ++c) h[c] = (f16)ra[j][c];
                *(v4h*)&As[buf][mm][c4 * 4] = h;
            }
        }
#pragma unroll
        for (int c = 0; c < 4; ++c) {
            v4h h;
#pragma unroll
            for (int j = 0; j < 4; ++j) h[j] = (f16)rb[j][c];
            *(v4h*)&Bs[buf][cb + c][kb] = h;
        }
    };

    v8f acc[2][4];
#pragma unroll
    for (int r = 0; r < 2; ++r)
#pragma unroll
        for (int c = 0; c < 4; ++c)
#pragma unroll
            for (int i = 0; i < 8; ++i) acc[r][c][i] = 0.0f;

    const int niter = K / BK;

    // prologue: stage tile 0 into buffer 0
    {
        v4f ra[4], rb[4];
        loadA(0, ra);
        loadB(0, rb);
        storeTile(ra, rb, 0);
    }
    __syncthreads();

    for (int it = 0; it < niter; ++it) {
        const int cur = it & 1;
        const bool hasNext = (it + 1 < niter);

        // 1) issue next tile's global loads (latency hidden behind WMMA below)
        v4f ra[4], rb[4];
        if (hasNext) {
            const int kn = (it + 1) * BK;
            loadA(kn, ra);
            loadB(kn, rb);
            if (it + 2 < niter) {   // distance-2 hint on the big streamed operand
                const int kp = (it + 2) * BK;
                if (TRANSA)
                    __builtin_prefetch(A + (size_t)(kp + kb) * lda + m0 + cb, 0, 3);
                else
                    __builtin_prefetch(A + (size_t)(m0 + (tid >> 3)) * lda + kp, 0, 3);
            }
        }

        // 2) fragments from current buffer (16-bit WMMA VGPR layout):
        //    A 16x32: lanes 0-15 (M=l16) hold K 0-7 & 16-23; lanes 16-31 K 8-15 & 24-31
        v16h afrag[2];
#pragma unroll
        for (int r = 0; r < 2; ++r) {
            int mrow = wm * 32 + r * 16 + l16;
            v8h lo = *(const v8h*)&As[cur][mrow][khalf * 8];
            v8h hi = *(const v8h*)&As[cur][mrow][khalf * 8 + 16];
#pragma unroll
            for (int i = 0; i < 8; ++i) { afrag[r][i] = lo[i]; afrag[r][i + 8] = hi[i]; }
        }
        //    B 32x16: lane holds column n=l16, K = khalf*8 + 0..7 & 16 + khalf*8 + 0..7 (2 per VGPR)
        v16h bfrag[4];
#pragma unroll
        for (int c = 0; c < 4; ++c) {
            int ncol = wn * 64 + c * 16 + l16;
            v8h b0 = *(const v8h*)&Bs[cur][ncol][khalf * 8];
            v8h b1 = *(const v8h*)&Bs[cur][ncol][khalf * 8 + 16];
#pragma unroll
            for (int i = 0; i < 8; ++i) { bfrag[c][i] = b0[i]; bfrag[c][i + 8] = b1[i]; }
        }

        // 3) 8 WMMAs per wave per K-step
#pragma unroll
        for (int r = 0; r < 2; ++r)
#pragma unroll
            for (int c = 0; c < 4; ++c)
                acc[r][c] = __builtin_amdgcn_wmma_f32_16x16x32_f16(
                    false, afrag[r], false, bfrag[c],
                    (short)0, acc[r][c], false, false);

        // 4) convert + store next tile into the other buffer; single barrier
        if (hasNext) storeTile(ra, rb, cur ^ 1);
        __syncthreads();
    }

    // epilogue: C/D layout -> lanes 0-15: M=v, lanes 16-31: M=8+v (coalesced per half-wave)
#pragma unroll
    for (int r = 0; r < 2; ++r) {
#pragma unroll
        for (int c = 0; c < 4; ++c) {
            int row0 = m0 + wm * 32 + r * 16 + khalf * 8;
            int col  = n0 + wn * 64 + c * 16 + l16;
#pragma unroll
            for (int v = 0; v < 8; ++v)
                C[(size_t)(row0 + v) * ldc + col] = acc[r][c][v];
        }
    }
}

// out[i] = (c0*x + c1*a + c2*b) * (diag ? diag[row] : 1),  row = i >> 8 (F=256)
__global__ __launch_bounds__(256)
void combine_poly(const float* __restrict__ x, const float* __restrict__ a,
                  const float* __restrict__ b, const float* __restrict__ par,
                  const float* __restrict__ diag, float* __restrict__ out,
                  int pbase, size_t total)
{
    size_t i = (size_t)blockIdx.x * blockDim.x + threadIdx.x;
    if (i >= total) return;
    float c0 = par[pbase + 0], c1 = par[pbase + 1], c2 = par[pbase + 2];
    float v = c0 * x[i] + c1 * a[i] + c2 * b[i];
    if (diag) v *= diag[i >> 8];      // F == 256
    out[i] = v;
}

extern "C" void kernel_launch(void* const* d_in, const int* in_sizes, int n_in,
                              void* d_out, int out_size, void* d_ws, size_t ws_size,
                              hipStream_t stream)
{
    const float* features = (const float*)d_in[0];   // [N,F]
    const float* theta    = (const float*)d_in[1];   // [N,N]
    const float* par      = (const float*)d_in[2];   // [6]
    const float* diag     = (const float*)d_in[3];   // [N]
    const float* weight   = (const float*)d_in[4];   // [F,OUT]
    float* out = (float*)d_out;                      // [N,OUT]

    const int N = 16384, F = 256, OUTC = 128;
    const size_t S = (size_t)N * F;                  // 4M floats per slot
    float* t1 = (float*)d_ws;                        // slot0
    float* t2 = t1 + S;                              // slot1
    float* y  = t2 + S;                              // slot2   (48 MB total)

    dim3 blk(256);
    dim3 grid_nf(N / BM, F / BN);                    // 128 x 2
    dim3 grid_no(N / BM, OUTC / BN);                 // 128 x 1
    unsigned eb = (unsigned)((S + 255) / 256);

    // t1 = Theta^T X
    gemm_f16_wmma<true ><<<grid_nf, blk, 0, stream>>>(theta, features, t1, N, F, N, N, F, F);
    // t2 = Theta^T t1
    gemm_f16_wmma<true ><<<grid_nf, blk, 0, stream>>>(theta, t1, t2, N, F, N, N, F, F);
    // y = diag * (p3 X + p4 t1 + p5 t2)
    combine_poly<<<eb, blk, 0, stream>>>(features, t1, t2, par, diag, y, 3, S);
    // p1 = Theta y   (reuse slot0)
    gemm_f16_wmma<false><<<grid_nf, blk, 0, stream>>>(theta, y, t1, N, F, N, N, F, F);
    // p2 = Theta p1  (reuse slot1)
    gemm_f16_wmma<false><<<grid_nf, blk, 0, stream>>>(theta, t1, t2, N, F, N, N, F, F);
    // z = p0 y + p1 p1v + p2 p2v  (in place over slot0; pure elementwise)
    combine_poly<<<eb, blk, 0, stream>>>(y, t1, t2, par, nullptr, t1, 0, S);
    // out = z W
    gemm_f16_wmma<false><<<grid_no, blk, 0, stream>>>(t1, weight, out, N, OUTC, F, F, OUTC, OUTC);
}
